// StackedLinear_63814624084079
// MI455X (gfx1250) — matrix operand order repaired and checked by reading, not compiled, not run on hardware
//
#include <hip/hip_runtime.h>
#include <hip/hip_bf16.h>
#include <math.h>

// ---------------------------------------------------------------------------
// StackedLinear (MoE-style selected linear) with LSQ-quantized weights.
//   out[b, o] = sum_i Wq[idx[b], o, i] * x[b, i] + bias[idx[b], o]
//   Wq = round(clip(W/alpha, -128, 127)) * alpha,  alpha = exp(log_alpha) per row
//
// Strategy (MI455X / gfx1250, wave32, WMMA):
//   * alpha factors out of the K-reduction -> pre-quantize q=round(clip(.)) to
//     f16 (exact: |q| <= 128), do f16 WMMA with f32 accum, scale by alpha and
//     add bias in the epilogue.
//   * bucket samples by expert (atomic histogram), then grouped GEMM:
//     each wave computes a 16-sample x 32-output tile with
//     V_WMMA_F32_16X16X32_F16 over K=1024 (32 K-steps, 2 N-tiles).
// ---------------------------------------------------------------------------

#define COUNT 8
#define OUT_F 32
#define IN_F  1024
#define BATCH 8192

typedef __attribute__((ext_vector_type(16))) _Float16 v16h;
typedef __attribute__((ext_vector_type(8)))  float    v8f;
typedef __attribute__((ext_vector_type(4)))  float    v4f;

// Workspace layout (bytes):
#define WS_QW_OFF     0u                      // 256*1024 f16 = 524288
#define WS_ALPHA_OFF  524288u                 // 256 f32     = 1024
#define WS_COUNT_OFF  525312u                 // 8 int       = 32 (padded)
#define WS_LIST_OFF   525344u                 // 8*8192 int  = 262144

// ---------------------------------------------------------------------------
// Kernel 1: quantize weights to f16 integers, compute alpha per row.
//   grid = 256 blocks (one per output row), block = 256 threads, 4 cols each.
// ---------------------------------------------------------------------------
__global__ __launch_bounds__(256) void quantize_weights_kernel(
    const float* __restrict__ weight,      // [256, 1024]
    const float* __restrict__ log_alpha,   // [256]
    _Float16* __restrict__ qw,             // [256, 1024] f16 out
    float* __restrict__ alpha_out)         // [256] f32 out
{
  const int row = blockIdx.x;
  const float alpha = __expf(log_alpha[row]);
  if (threadIdx.x == 0) alpha_out[row] = alpha;
  const float inv_alpha = 1.0f / alpha;

  const int c0 = threadIdx.x * 4;
  const float* wrow = weight + (size_t)row * IN_F;
  _Float16* qrow = qw + (size_t)row * IN_F;
#pragma unroll
  for (int j = 0; j < 4; ++j) {
    float ws = wrow[c0 + j] * inv_alpha;
    ws = fminf(fmaxf(ws, -128.0f), 127.0f);
    qrow[c0 + j] = (_Float16)rintf(ws);   // integer in [-128,127]: exact in f16
  }
}

// ---------------------------------------------------------------------------
// Kernel 2a: zero the expert histogram.
// ---------------------------------------------------------------------------
__global__ void zero_counts_kernel(int* __restrict__ counts) {
  if (threadIdx.x < COUNT) counts[threadIdx.x] = 0;
}

// ---------------------------------------------------------------------------
// Kernel 2b: bucket sample ids by expert.
// ---------------------------------------------------------------------------
__global__ __launch_bounds__(256) void bucket_kernel(
    const long long* __restrict__ ls_indices,  // [B] int64
    int* __restrict__ counts,                  // [8]
    int* __restrict__ lists)                   // [8][B]
{
  const int i = blockIdx.x * blockDim.x + threadIdx.x;
  if (i >= BATCH) return;
  const int e = (int)ls_indices[i];
  const int pos = atomicAdd(&counts[e], 1);
  lists[e * BATCH + pos] = i;
}

// ---------------------------------------------------------------------------
// Kernel 3: grouped GEMM per expert.
//   grid = (128, 8): blockIdx.y = expert, 4 waves/block, 16 samples/wave.
//   Wave tile: M=16 samples, N=32 outputs (two 16x16 WMMA tiles), K=1024.
//
// VGPR fragment layouts (wave32; A/B K order contiguous as used here, C/D per CDNA5 ISA 7.12.2):
//   A (16x32 f16):  lanes 0-15 -> M=lane,   halfs j: K=k0+j   (j=0..15)
//                   lanes 16-31 -> M=lane-16, halfs j: K=k0+16+j
//   B (32x16 f16):  lanes 0-15 -> N=lane,   halfs j: K=k0+j   (j=0..15)
//                   lanes 16-31 -> N=lane-16, halfs j: K=k0+16+j
//                   -> 16 K-contiguous f16 per lane: one 32B vector load,
//                      since qw rows are K-major.
//   C/D (16x16 f32): VGPR r, lanes 0-15 -> (M=r, N=lane),
//                            lanes 16-31 -> (M=8+r, N=lane-16)
// ---------------------------------------------------------------------------
__global__ __launch_bounds__(128) void moe_gemm_kernel(
    const float* __restrict__ x,          // [B, 1024]
    const _Float16* __restrict__ qw,      // [8*32, 1024] f16
    const float* __restrict__ alpha,      // [256]
    const float* __restrict__ bias,       // [256]
    const int* __restrict__ counts,       // [8]
    const int* __restrict__ lists,        // [8][B]
    float* __restrict__ out)              // [B, 32]
{
  const int lane    = threadIdx.x & 31;
  const int wave    = threadIdx.x >> 5;
  const int e       = blockIdx.y;
  const int cnt     = counts[e];
  const int base    = blockIdx.x * 64 + wave * 16;
  if (base >= cnt) return;

  const int halfSel = lane >> 4;     // 0: lanes 0-15, 1: lanes 16-31
  const int mrow    = lane & 15;

  const int* list = lists + e * BATCH;

  // Sample this lane loads A-rows for (clamped for the ragged tail).
  int li = base + mrow;
  if (li >= cnt) li = cnt - 1;
  const int sample = list[li];
  const float* xrow = x + (size_t)sample * IN_F;

  // B-matrix pointers: column n, 16 K-contiguous halfs starting at k0+16*halfSel.
  const int n0 = mrow;               // tile 0 column
  const int n1 = 16 + mrow;          // tile 1 column
  const _Float16* wrow0 = qw + ((size_t)(e * OUT_F + n0)) * IN_F + 16 * halfSel;
  const _Float16* wrow1 = qw + ((size_t)(e * OUT_F + n1)) * IN_F + 16 * halfSel;

  v8f acc0 = {};
  v8f acc1 = {};

  const int ka_base = 16 * halfSel;  // A-fragment K offset for this lane half

#pragma unroll 4
  for (int k0 = 0; k0 < IN_F; k0 += 32) {
    // Prefetch next x chunk (emits global_prefetch_b8).
    __builtin_prefetch(xrow + k0 + 32 + ka_base, 0, 1);

    // ---- A fragment: 16 consecutive f32 -> f16 ---------------------------
    const int ka = k0 + ka_base;
    v4f a0 = *(const v4f*)(xrow + ka);
    v4f a1 = *(const v4f*)(xrow + ka + 4);
    v4f a2 = *(const v4f*)(xrow + ka + 8);
    v4f a3 = *(const v4f*)(xrow + ka + 12);
    v16h A;
#pragma unroll
    for (int j = 0; j < 4; ++j) {
      A[j]      = (_Float16)a0[j];
      A[4 + j]  = (_Float16)a1[j];
      A[8 + j]  = (_Float16)a2[j];
      A[12 + j] = (_Float16)a3[j];
    }

    // ---- B fragments: 16 K-contiguous f16 = one 32B load each ------------
    v16h B0 = *(const v16h*)(wrow0 + k0);
    v16h B1 = *(const v16h*)(wrow1 + k0);

    // ---- D = A x B + C ----------------------------------------------------
    acc0 = __builtin_amdgcn_wmma_f32_16x16x32_f16(
        /*neg_a=*/false, A, /*neg_b=*/false, B0,
        /*c_mod=*/(short)0, acc0, /*reuse_a=*/false, /*reuse_b=*/false);
    acc1 = __builtin_amdgcn_wmma_f32_16x16x32_f16(
        /*neg_a=*/false, A, /*neg_b=*/false, B1,
        /*c_mod=*/(short)0, acc1, /*reuse_a=*/false, /*reuse_b=*/false);
  }

  // ---- Epilogue: out[sample, n] = alpha[e*32+n]*acc + bias[e*32+n] --------
  const float al0 = alpha[e * OUT_F + n0];
  const float bi0 = bias [e * OUT_F + n0];
  const float al1 = alpha[e * OUT_F + n1];
  const float bi1 = bias [e * OUT_F + n1];

#pragma unroll
  for (int r = 0; r < 8; ++r) {
    const int M = r + 8 * halfSel;          // C/D row held by this lane/VGPR
    const int li2 = base + M;
    if (li2 < cnt) {
      const int s = list[li2];
      out[(size_t)s * OUT_F + n0] = al0 * acc0[r] + bi0;
      out[(size_t)s * OUT_F + n1] = al1 * acc1[r] + bi1;
    }
  }
}

// ---------------------------------------------------------------------------
// Host-side launcher.
// Inputs: 0=x [B,1024] f32, 1=ls_indices [B] i64, 2=weight [256,1024] f32,
//         3=bias [256] f32, 4=log_alpha_w [256] f32.  Out: [B,32] f32.
// ---------------------------------------------------------------------------
extern "C" void kernel_launch(void* const* d_in, const int* in_sizes, int n_in,
                              void* d_out, int out_size, void* d_ws, size_t ws_size,
                              hipStream_t stream) {
  const float*     x      = (const float*)d_in[0];
  const long long* lsi    = (const long long*)d_in[1];
  const float*     weight = (const float*)d_in[2];
  const float*     bias   = (const float*)d_in[3];
  const float*     lalpha = (const float*)d_in[4];
  float*           out    = (float*)d_out;

  char* ws = (char*)d_ws;
  _Float16* qw     = (_Float16*)(ws + WS_QW_OFF);
  float*    alpha  = (float*)   (ws + WS_ALPHA_OFF);
  int*      counts = (int*)     (ws + WS_COUNT_OFF);
  int*      lists  = (int*)     (ws + WS_LIST_OFF);

  // 1) Quantize weights (f16 integer mantissa) + alpha.
  quantize_weights_kernel<<<dim3(COUNT * OUT_F), dim3(256), 0, stream>>>(
      weight, lalpha, qw, alpha);

  // 2) Bucket samples by expert.
  zero_counts_kernel<<<dim3(1), dim3(32), 0, stream>>>(counts);
  bucket_kernel<<<dim3(BATCH / 256), dim3(256), 0, stream>>>(lsi, counts, lists);

  // 3) Grouped WMMA GEMM: grid covers worst case (all samples on one expert).
  moe_gemm_kernel<<<dim3(BATCH / 64, COUNT), dim3(128), 0, stream>>>(
      x, qw, alpha, bias, counts, lists, out);
}
